// Attention_5849745457789
// MI455X (gfx1250) — matrix operand order repaired and checked by reading, not compiled, not run on hardware
//
#include <hip/hip_runtime.h>
#include <hip/hip_bf16.h>
#include <float.h>

// ---------------------------------------------------------------------------
// Problem constants (from reference): B=4, N=2048, D=1024, H=16, DH=64
// ---------------------------------------------------------------------------
#define BB     4
#define NN     2048
#define DD     1024
#define HH     16
#define DHH    64
#define INNER  (HH * DHH)      // 1024
#define MAXP   256
#define TOPK   64
#define QSCALE 0.125f          // 64^-0.5

typedef __bf16 bf16;
typedef __attribute__((ext_vector_type(16))) __bf16 v16bf;
typedef __attribute__((ext_vector_type(8)))  float  v8f;

// ---------------------------------------------------------------------------
// Workspace layout (bytes). Total ~88 MB.
// ---------------------------------------------------------------------------
static const size_t WS_Q   = 0;                         // (B,H,N,DH) bf16, pre-scaled
static const size_t WS_K   = WS_Q   + 16777216;         // (B,H,N,DH) bf16
static const size_t WS_V   = WS_K   + 16777216;         // (B,H,DH,N) bf16 (transposed)
static const size_t WS_X   = WS_V   + 16777216;         // x bf16
static const size_t WS_WQ  = WS_X   + 16777216;         // WqT  bf16 (1024c x 1024k)
static const size_t WS_WKV = WS_WQ  + 2097152;          // WkvT bf16 (2048c x 1024k)
static const size_t WS_WO  = WS_WKV + 4194304;          // WoT  bf16 (1024c x 1024k)
static const size_t WS_RE  = WS_WO  + 2097152;          // rel_emb bf16 (256x64)
static const size_t WS_AO  = WS_RE  + 32768;            // attn out bf16 (B*N x 1024)

// ---------------------------------------------------------------------------
// CDNA5 async global->LDS copy (ASYNCcnt-tracked DMA, ISA 10.7/15.18).
// Generic LDS pointer low 32 bits == workgroup-relative LDS address (ISA 10.2).
// ---------------------------------------------------------------------------
__device__ inline void async_g2l_b128(const void* gptr, void* lptr) {
  unsigned lds_addr = (unsigned)(unsigned long long)lptr;
  asm volatile("global_load_async_to_lds_b128 %0, %1, off"
               :: "v"(lds_addr), "v"(gptr)
               : "memory");
}
__device__ inline void wait_async0() {
  asm volatile("s_wait_asynccnt 0x0" ::: "memory");
}

// ---------------------------------------------------------------------------
// WMMA fragment helpers (v_wmma_f32_16x16x32_bf16), ISA 7.12.2 layouts:
//  A (16x32 MxK): lane&15 = row; two 16B loads at row*ld + half*8 and +16
//  B (32x16 KxN): lane&15 = col; two 16B loads at col base + half*8 and +16
//  C (16x16 f32): VGPR r -> row r + 8*(lane>>4), col = lane&15
// ---------------------------------------------------------------------------
union fragU { uint4 u[2]; v16bf v; };

__device__ inline v8f zero_v8f() {
  v8f z;
#pragma unroll
  for (int i = 0; i < 8; ++i) z[i] = 0.f;
  return z;
}

__device__ inline v8f wmma_bf16(v16bf a, v16bf b, v8f c) {
  return __builtin_amdgcn_wmma_f32_16x16x32_bf16(false, a, false, b, (short)0, c, false, false);
}

__device__ inline v16bf load_a_frag(const bf16* tile, int ld, int lane) {
  const bf16* rp = tile + (size_t)(lane & 15) * ld + (lane >> 4) * 8;
  fragU f;
  f.u[0] = *(const uint4*)(rp);
  f.u[1] = *(const uint4*)(rp + 16);
  return f.v;
}

__device__ inline v16bf load_b_frag(const bf16* colptr, int lane) {
  const bf16* cp = colptr + (lane >> 4) * 8;
  fragU f;
  f.u[0] = *(const uint4*)(cp);
  f.u[1] = *(const uint4*)(cp + 16);
  return f.v;
}

// wave32 reductions
__device__ inline float wred_max(float v) {
#pragma unroll
  for (int o = 16; o > 0; o >>= 1) v = fmaxf(v, __shfl_xor(v, o, 32));
  return v;
}
__device__ inline float wred_sum(float v) {
#pragma unroll
  for (int o = 16; o > 0; o >>= 1) v += __shfl_xor(v, o, 32);
  return v;
}
__device__ inline int wred_sumi(int v) {
#pragma unroll
  for (int o = 16; o > 0; o >>= 1) v += __shfl_xor(v, o, 32);
  return v;
}

// order-preserving float <-> uint mapping (for exact k-th largest)
__device__ inline unsigned f2ord(float f) {
  unsigned u = __float_as_uint(f);
  return (u & 0x80000000u) ? ~u : (u | 0x80000000u);
}
__device__ inline float ord2f(unsigned m) {
  unsigned u = (m & 0x80000000u) ? (m & 0x7fffffffu) : ~m;
  return __uint_as_float(u);
}

__device__ inline unsigned pack2bf(float a, float b) {
  union { bf16 h[2]; unsigned u; } o;
  o.h[0] = (bf16)a; o.h[1] = (bf16)b;
  return o.u;
}

// ---------------------------------------------------------------------------
// Kernel 0a: fp32 -> bf16 convert, 4 elems/thread (n % 4 == 0)
// ---------------------------------------------------------------------------
__global__ void cvt_f32_bf16_v4(const float* __restrict__ s, bf16* __restrict__ d, int n4) {
  for (int i = blockIdx.x * blockDim.x + threadIdx.x; i < n4; i += gridDim.x * blockDim.x) {
    float4 f = ((const float4*)s)[i];
    union { bf16 h[4]; uint2 u; } o;
    o.h[0] = (bf16)f.x; o.h[1] = (bf16)f.y; o.h[2] = (bf16)f.z; o.h[3] = (bf16)f.w;
    ((uint2*)d)[i] = o.u;
  }
}

// Kernel 0b: fp32 -> bf16 transposed convert: d[c*rows + k] = s[k*cols + c]
__global__ void cvt_f32_bf16_T(const float* __restrict__ s, bf16* __restrict__ d,
                               int rows, int cols) {
  int n = rows * cols;
  for (int i = blockIdx.x * blockDim.x + threadIdx.x; i < n; i += gridDim.x * blockDim.x) {
    int k = i / cols, c = i - k * cols;       // reads coalesced over c
    d[(size_t)c * rows + k] = (bf16)s[i];
  }
}

// ---------------------------------------------------------------------------
// Kernel 1: fused QKV projection with async double-buffered staging.
// grid (64, 48), 256 threads (8 waves). Block tile 128x64, K-chunks of 64.
// ---------------------------------------------------------------------------
__global__ __launch_bounds__(256) void qkv_gemm_kernel(
    const bf16* __restrict__ xB, const bf16* __restrict__ wqT, const bf16* __restrict__ wkvT,
    const float* __restrict__ bq, const float* __restrict__ bkv,
    bf16* __restrict__ qB, bf16* __restrict__ kB, bf16* __restrict__ vB) {
  __shared__ bf16 sX[2][128 * 64];   // 2 x 16 KB
  __shared__ bf16 sW[2][64 * 64];    // 2 x 8 KB   (sW[col*64 + k])
  const int m0 = blockIdx.x * 128, c0 = blockIdx.y * 64;
  const int tid = threadIdx.x, lane = tid & 31, wv = tid >> 5;
  const int half = lane >> 4, coll = lane & 15;

  // whole 64-col tile lies in one weight segment (c0 multiple of 64)
  const bf16* wbase = (c0 < INNER) ? (wqT + (size_t)c0 * DD)
                                   : (wkvT + (size_t)(c0 - INNER) * DD);
  // per-thread staging coordinates: row = tid>>3 (+32*i), 16B chunk = tid&7
  const int srow = tid >> 3, sq = (tid & 7) * 8;
  const bf16* xsrc = xB + (size_t)(m0 + srow) * DD + sq;
  const bf16* wsrc = wbase + (size_t)srow * DD + sq;

  v8f acc[4];
#pragma unroll
  for (int ct = 0; ct < 4; ++ct) acc[ct] = zero_v8f();

#define QKV_ISSUE(kc_)                                                          \
  {                                                                             \
    int _b = (kc_) & 1;                                                         \
    bf16* xd = &sX[_b][srow * 64 + sq];                                         \
    bf16* wd = &sW[_b][srow * 64 + sq];                                         \
    _Pragma("unroll")                                                           \
    for (int i = 0; i < 4; ++i)                                                 \
      async_g2l_b128(xsrc + (size_t)(32 * i) * DD + (kc_) * 64, xd + (32 * i) * 64); \
    _Pragma("unroll")                                                           \
    for (int i = 0; i < 2; ++i)                                                 \
      async_g2l_b128(wsrc + (size_t)(32 * i) * DD + (kc_) * 64, wd + (32 * i) * 64); \
  }

  QKV_ISSUE(0);
  wait_async0();
  __syncthreads();

  for (int kc = 0; kc < 16; ++kc) {
    if (kc + 1 < 16) QKV_ISSUE(kc + 1);           // DMA next chunk during compute
    const bf16* sXc = &sX[kc & 1][0];
    const bf16* sWc = &sW[kc & 1][0];
#pragma unroll
    for (int kk = 0; kk < 64; kk += 32) {
      v16bf a  = load_a_frag(sXc + wv * (16 * 64) + kk, 64, lane);
      v16bf b0 = load_b_frag(sWc + (0 * 16 + coll) * 64 + kk, lane);
      v16bf b1 = load_b_frag(sWc + (1 * 16 + coll) * 64 + kk, lane);
      v16bf b2 = load_b_frag(sWc + (2 * 16 + coll) * 64 + kk, lane);
      v16bf b3 = load_b_frag(sWc + (3 * 16 + coll) * 64 + kk, lane);
      acc[0] = wmma_bf16(a, b0, acc[0]);
      acc[1] = wmma_bf16(a, b1, acc[1]);
      acc[2] = wmma_bf16(a, b2, acc[2]);
      acc[3] = wmma_bf16(a, b3, acc[3]);
    }
    wait_async0();
    __syncthreads();
  }
#undef QKV_ISSUE

#pragma unroll
  for (int ct = 0; ct < 4; ++ct) {
#pragma unroll
    for (int r = 0; r < 8; ++r) {
      int m = m0 + wv * 16 + r + 8 * half;
      int c = c0 + ct * 16 + coll;
      float bias = (c < INNER) ? bq[c] : bkv[c - INNER];
      float val = acc[ct][r] + bias;
      int b_ = m >> 11, n = m & 2047;
      if (c < INNER) {                     // Q, pre-scaled, (B,H,N,DH)
        int hh = c >> 6, dh = c & 63;
        qB[(((size_t)(b_ * HH + hh) * NN) + n) * DHH + dh] = (bf16)(val * QSCALE);
      } else if (c < 2 * INNER) {          // K, (B,H,N,DH)
        int cc = c - INNER, hh = cc >> 6, dh = cc & 63;
        kB[(((size_t)(b_ * HH + hh) * NN) + n) * DHH + dh] = (bf16)val;
      } else {                             // V, transposed (B,H,DH,N)
        int cc = c - 2 * INNER, hh = cc >> 6, dh = cc & 63;
        vB[(((size_t)(b_ * HH + hh) * DHH) + dh) * NN + n] = (bf16)val;
      }
    }
  }
}

// ---------------------------------------------------------------------------
// Kernel 2: attention with exact top-k + rel-pos bias + gated softmax + P@V.
// One block per (b, h, 16-query tile); 512 threads = 16 waves.
// Dynamic LDS (203,008 B): sDots f32 16x2048 | sQ | pad | union(sQE/sP) | sOut
// ---------------------------------------------------------------------------
#define ATTN_SMEM 203008

__global__ __launch_bounds__(512) void attn_topk_kernel(
    const bf16* __restrict__ qB, const bf16* __restrict__ kB, const bf16* __restrict__ vB,
    const bf16* __restrict__ reB, const float* __restrict__ gating,
    const unsigned char* __restrict__ maskp, bf16* __restrict__ aoB) {
  extern __shared__ char smem[];
  float* sDots = (float*)(smem);
  bf16*  sQ    = (bf16*)(smem + 131072);
  float* sQE   = (float*)(smem + 133376);
  bf16*  sP    = (bf16*)(smem + 133376);
  float* sOut  = (float*)(smem + 198912);

  const int bid = blockIdx.x;
  const int it  = bid & 127;
  const int h   = (bid >> 7) & 15;
  const int b   = bid >> 11;
  const int i0  = it * 16;
  const int tid = threadIdx.x, lane = tid & 31, wv = tid >> 5;
  const int half = lane >> 4, coll = lane & 15;

  const size_t headOff = (size_t)(b * HH + h) * NN * DHH;
  const bf16* kHead  = kB + headOff;
  const bf16* vHeadT = vB + headOff;      // (DH, N) transposed

  for (int idx = tid; idx < 16 * DHH; idx += blockDim.x) {
    int r = idx >> 6, d = idx & 63;
    sQ[r * DHH + d] = qB[headOff + (size_t)(i0 + r) * DHH + d];
    sOut[idx] = 0.f;
  }
  __syncthreads();

  // ---- qE = Qs @ rel_emb^T (16x256, K=64); wave -> 16 rel positions ----
  {
    int r0 = wv * 16;
    v8f acc = zero_v8f();
#pragma unroll
    for (int kk = 0; kk < 2; ++kk) {
      v16bf a  = load_a_frag(sQ + kk * 32, DHH, lane);
      v16bf bb = load_b_frag(reB + (size_t)(r0 + coll) * DHH + kk * 32, lane);
      acc = wmma_bf16(a, bb, acc);
    }
#pragma unroll
    for (int r = 0; r < 8; ++r)
      sQE[(r + 8 * half) * MAXP + r0 + coll] = acc[r];
  }
  __syncthreads();

  // ---- dots = Qs @ K^T + pos_attn, masked; wave w -> cols [128w,128w+128) ----
  {
    v16bf a0 = load_a_frag(sQ + 0,  DHH, lane);
    v16bf a1 = load_a_frag(sQ + 32, DHH, lane);
#pragma unroll
    for (int t = 0; t < 8; ++t) {
      int j0 = wv * 128 + t * 16;
      int jg = j0 + coll;
      v8f acc = zero_v8f();
      v16bf b0 = load_b_frag(kHead + (size_t)jg * DHH + 0,  lane);
      v16bf b1 = load_b_frag(kHead + (size_t)jg * DHH + 32, lane);
      acc = wmma_bf16(a0, b0, acc);
      acc = wmma_bf16(a1, b1, acc);
      bool mj = maskp[b * NN + jg] != 0;
#pragma unroll
      for (int r = 0; r < 8; ++r) {
        int row = r + 8 * half;
        int ig = i0 + row;
        int dist = ig - jg + MAXP;
        dist = dist < 0 ? 0 : (dist > MAXP - 1 ? MAXP - 1 : dist);
        float val = acc[r] + sQE[row * MAXP + dist];
        if (!mj || maskp[b * NN + ig] == 0) val = -FLT_MAX;
        sDots[row * NN + jg] = val;
      }
    }
  }
  __syncthreads();

  // ---- per-row exact top-k (binary search over ordered-uint), softmax, P ----
  {
    const float* rowp = sDots + wv * NN;
    float mx = -FLT_MAX;
    for (int j = lane; j < NN; j += 32) mx = fmaxf(mx, rowp[j]);
    mx = wred_max(mx);

    unsigned lo = 0u, hi = f2ord(mx);
    while (lo < hi) {
      unsigned span = hi - lo;
      unsigned mid = lo + (span >> 1) + (span & 1u);
      int cnt = 0;
      for (int j = lane; j < NN; j += 32) cnt += (f2ord(rowp[j]) >= mid) ? 1 : 0;
      cnt = wred_sumi(cnt);
      if (cnt >= TOPK) lo = mid; else hi = mid - 1;
    }
    float vk = ord2f(lo);

    float s = 0.f;
    for (int j = lane; j < NN; j += 32) {
      float x = rowp[j];
      if (x >= vk) s += __expf(x - mx);
    }
    s = wred_sum(s);
    float coef = gating[b * NN + i0 + wv] / s;
    __syncthreads();            // sQE dead; union region becomes sP
    for (int j2 = lane; j2 < NN / 2; j2 += 32) {
      float2 x = *(const float2*)(rowp + j2 * 2);
      float p0 = (x.x >= vk) ? __expf(x.x - mx) * coef : 0.f;
      float p1 = (x.y >= vk) ? __expf(x.y - mx) * coef : 0.f;
      *(unsigned*)(sP + wv * NN + j2 * 2) = pack2bf(p0, p1);
    }
  }
  __syncthreads();

  // ---- out = P @ V^T (16x64, K=2048). wave -> (ct = wv&3, kq = wv>>2) ----
  {
    int ct = wv & 3, kq = wv >> 2;
    v8f acc = zero_v8f();
#pragma unroll 4
    for (int ks = 0; ks < 16; ++ks) {
      int kb = kq * 512 + ks * 32;
      v16bf a  = load_a_frag(sP + kb, NN, lane);
      v16bf bb = load_b_frag(vHeadT + (size_t)(ct * 16 + coll) * NN + kb, lane);
      acc = wmma_bf16(a, bb, acc);
    }
#pragma unroll
    for (int r = 0; r < 8; ++r)
      atomicAdd(&sOut[(r + 8 * half) * DHH + ct * 16 + coll], acc[r]);
  }
  __syncthreads();

  for (int idx = tid; idx < 16 * DHH / 2; idx += blockDim.x) {
    int r = idx >> 5, d2 = idx & 31;
    float o0 = sOut[r * DHH + d2 * 2], o1 = sOut[r * DHH + d2 * 2 + 1];
    *(unsigned*)(aoB + ((size_t)(b * NN + i0 + r)) * INNER + h * DHH + d2 * 2) =
        pack2bf(o0, o1);
  }
}

// ---------------------------------------------------------------------------
// Kernel 3: output projection with async double-buffered staging.
// grid (64, 16), 256 threads. Block tile 128x64, K-chunks of 64. fp32 out.
// ---------------------------------------------------------------------------
__global__ __launch_bounds__(256) void out_gemm_kernel(
    const bf16* __restrict__ aoB, const bf16* __restrict__ woT,
    const float* __restrict__ bo, float* __restrict__ out) {
  __shared__ bf16 sX[2][128 * 64];
  __shared__ bf16 sW[2][64 * 64];
  const int m0 = blockIdx.x * 128, c0 = blockIdx.y * 64;
  const int tid = threadIdx.x, lane = tid & 31, wv = tid >> 5;
  const int half = lane >> 4, coll = lane & 15;

  const int srow = tid >> 3, sq = (tid & 7) * 8;
  const bf16* xsrc = aoB + (size_t)(m0 + srow) * INNER + sq;
  const bf16* wsrc = woT + (size_t)(c0 + srow) * INNER + sq;

  v8f acc[4];
#pragma unroll
  for (int ct = 0; ct < 4; ++ct) acc[ct] = zero_v8f();

#define OUT_ISSUE(kc_)                                                          \
  {                                                                             \
    int _b = (kc_) & 1;                                                         \
    bf16* xd = &sX[_b][srow * 64 + sq];                                         \
    bf16* wd = &sW[_b][srow * 64 + sq];                                         \
    _Pragma("unroll")                                                           \
    for (int i = 0; i < 4; ++i)                                                 \
      async_g2l_b128(xsrc + (size_t)(32 * i) * INNER + (kc_) * 64, xd + (32 * i) * 64); \
    _Pragma("unroll")                                                           \
    for (int i = 0; i < 2; ++i)                                                 \
      async_g2l_b128(wsrc + (size_t)(32 * i) * INNER + (kc_) * 64, wd + (32 * i) * 64); \
  }

  OUT_ISSUE(0);
  wait_async0();
  __syncthreads();

  for (int kc = 0; kc < 16; ++kc) {
    if (kc + 1 < 16) OUT_ISSUE(kc + 1);
    const bf16* sXc = &sX[kc & 1][0];
    const bf16* sWc = &sW[kc & 1][0];
#pragma unroll
    for (int kk = 0; kk < 64; kk += 32) {
      v16bf a  = load_a_frag(sXc + wv * (16 * 64) + kk, 64, lane);
      v16bf b0 = load_b_frag(sWc + (0 * 16 + coll) * 64 + kk, lane);
      v16bf b1 = load_b_frag(sWc + (1 * 16 + coll) * 64 + kk, lane);
      v16bf b2 = load_b_frag(sWc + (2 * 16 + coll) * 64 + kk, lane);
      v16bf b3 = load_b_frag(sWc + (3 * 16 + coll) * 64 + kk, lane);
      acc[0] = wmma_bf16(a, b0, acc[0]);
      acc[1] = wmma_bf16(a, b1, acc[1]);
      acc[2] = wmma_bf16(a, b2, acc[2]);
      acc[3] = wmma_bf16(a, b3, acc[3]);
    }
    wait_async0();
    __syncthreads();
  }
#undef OUT_ISSUE

#pragma unroll
  for (int ct = 0; ct < 4; ++ct) {
#pragma unroll
    for (int r = 0; r < 8; ++r) {
      int m = m0 + wv * 16 + r + 8 * half;
      int c = c0 + ct * 16 + coll;
      out[(size_t)m * DD + c] = acc[ct][r] + bo[c];
    }
  }
}

// ---------------------------------------------------------------------------
// Host launcher
// ---------------------------------------------------------------------------
extern "C" void kernel_launch(void* const* d_in, const int* in_sizes, int n_in,
                              void* d_out, int out_size, void* d_ws, size_t ws_size,
                              hipStream_t stream) {
  const float* x      = (const float*)d_in[0];
  const float* gating = (const float*)d_in[1];
  const float* Wq     = (const float*)d_in[2];
  const float* bq     = (const float*)d_in[3];
  const float* Wkv    = (const float*)d_in[4];
  const float* bkv    = (const float*)d_in[5];
  const float* Wo     = (const float*)d_in[6];
  const float* bo     = (const float*)d_in[7];
  const float* rel    = (const float*)d_in[8];
  const unsigned char* mask = (const unsigned char*)d_in[9];
  float* out = (float*)d_out;

  char* ws = (char*)d_ws;
  bf16* qB   = (bf16*)(ws + WS_Q);
  bf16* kB   = (bf16*)(ws + WS_K);
  bf16* vB   = (bf16*)(ws + WS_V);
  bf16* xB   = (bf16*)(ws + WS_X);
  bf16* wqT  = (bf16*)(ws + WS_WQ);
  bf16* wkvT = (bf16*)(ws + WS_WKV);
  bf16* woT  = (bf16*)(ws + WS_WO);
  bf16* reB  = (bf16*)(ws + WS_RE);
  bf16* aoB  = (bf16*)(ws + WS_AO);

  (void)hipFuncSetAttribute(reinterpret_cast<const void*>(&attn_topk_kernel),
                            hipFuncAttributeMaxDynamicSharedMemorySize, ATTN_SMEM);

  {
    int n4 = (BB * NN * DD) / 4;
    int g = (n4 + 255) / 256; if (g > 16384) g = 16384;
    cvt_f32_bf16_v4<<<g, 256, 0, stream>>>(x, xB, n4);
    int nr = (MAXP * DHH) / 4;
    cvt_f32_bf16_v4<<<(nr + 255) / 256, 256, 0, stream>>>(rel, reB, nr);
    cvt_f32_bf16_T<<<4096, 256, 0, stream>>>(Wq,  wqT,  DD, INNER);
    cvt_f32_bf16_T<<<8192, 256, 0, stream>>>(Wkv, wkvT, DD, 2 * INNER);
    cvt_f32_bf16_T<<<4096, 256, 0, stream>>>(Wo,  woT,  INNER, DD);
  }

  // QKV projection: M=8192, N=3072, K=1024
  qkv_gemm_kernel<<<dim3(64, 48), 256, 0, stream>>>(xB, wqT, wkvT, bq, bkv, qB, kB, vB);

  // attention: B*H*(N/16) blocks
  attn_topk_kernel<<<BB * HH * (NN / 16), 512, ATTN_SMEM, stream>>>(
      qB, kB, vB, reB, gating, mask, aoB);

  // output projection: M=8192, N=1024, K=1024
  out_gemm_kernel<<<dim3(64, 16), 256, 0, stream>>>(aoB, woT, bo, out);
}
